// NeuromorphicLM_91302414778898
// MI455X (gfx1250) — matrix operand order repaired and checked by reading, compiled, not run on hardware
//
#include <hip/hip_runtime.h>
#include <hip/hip_bf16.h>

// NeuromorphicLM forward step for gfx1250 (MI455X).
// All matmuls via V_WMMA_F32_16X16X32_BF16 (bf16 mul, fp32 accumulate).
// A-side operands live as bf16 shadows in workspace (converted once by the
// producing kernel), so the GEMM hot loop is pure load->wmma. B-side (weights /
// emb) converts fp32->bf16 inline while staging through LDS: for the dominant
// 256x50257x1024 logits GEMM this keeps HBM traffic at the single-pass 206 MB
// minimum (23.3 TB/s roof ~= 11 us floor). Per-wave 32x32 macro-tile gives
// 16 WMMAs per barrier pair and a single-block (hence single-HBM-pass) read
// of each emb tile.

#define VOCAB 50257
#define DMODEL 1024
#define NBLK 8
#define NLAYER 2
#define DH 128
#define BS 256

typedef __attribute__((ext_vector_type(16))) __bf16        v16bf;
typedef __attribute__((ext_vector_type(16))) unsigned short v16us;
typedef __attribute__((ext_vector_type(8)))  float          v8f;

__device__ __forceinline__ unsigned short f2bf_rne(float f) {
  union { float f; unsigned u; } v; v.f = f;
  unsigned u = v.u;
  unsigned r = u + 0x7FFFu + ((u >> 16) & 1u);   // round-to-nearest-even
  return (unsigned short)(r >> 16);
}

__device__ __forceinline__ float sigmoidf_(float x) {
  return 1.0f / (1.0f + __expf(-x));
}

// ---------------------------------------------------------------------------
// Generic batched GEMM: C[b] (+)= A[b] @ B[b]^T        (M fixed = 256)
//   A: 256 x K bf16 (row major, lda), B: N x K fp32 (row major, ldb),
//   C: 256 x N fp32 (ldc), optional bf16 shadow Cb for chained GEMMs.
//   grid = (ceil(N/32), batch), block = 256 threads (8 waves, wave32).
//   Per wave: 32x32 macro-tile (2x2 WMMA tiles) -> 16 WMMAs per barrier pair.
// ---------------------------------------------------------------------------
template <bool ACC, bool WB16>
__global__ __launch_bounds__(256) void wmma_gemm_nt(
    const unsigned short* __restrict__ A, const float* __restrict__ Bw,
    float* __restrict__ C, unsigned short* __restrict__ Cb,
    int N, int K, int lda, int ldb, int ldc,
    long sA, long sB, long sC) {
  __shared__ __align__(16) unsigned short ldsB[32 * 128];  // 32 N-rows x 128 K

  const int bz = blockIdx.y;
  A  += (long)bz * sA;
  Bw += (long)bz * sB;
  C  += (long)bz * sC;
  if (WB16) Cb += (long)bz * sC;

  const int tid  = threadIdx.x;
  const int wave = tid >> 5;   // 0..7
  const int lane = tid & 31;
  const int half = lane >> 4;  // 0/1
  const int l16  = lane & 15;
  const int mBase = wave * 32;          // 8 waves x 32 rows = M = 256
  const int nTile = blockIdx.x * 32;

  // staging coords: thread stages 16 K-values of one B row
  const int sn  = tid >> 3;             // 0..31 : B row within tile
  const int sk0 = (tid & 7) * 16;       // 0..112
  int srow = nTile + sn;
  if (srow >= N) srow = N - 1;          // tail rows clamped; stores guarded
  const float* __restrict__ sBp = Bw + (long)srow * ldb + sk0;

  v8f acc00 = {}, acc01 = {}, acc10 = {}, acc11 = {};
  if (ACC) {
#pragma unroll
    for (int r = 0; r < 8; ++r) {
      const int row0 = mBase + half * 8 + r;
      const int row1 = row0 + 16;
      const int col0 = nTile + l16;
      const int col1 = col0 + 16;
      if (col0 < N) {
        acc00[r] = C[(long)row0 * ldc + col0];
        acc10[r] = C[(long)row1 * ldc + col0];
      }
      if (col1 < N) {
        acc01[r] = C[(long)row0 * ldc + col1];
        acc11[r] = C[(long)row1 * ldc + col1];
      }
    }
  }

  for (int kk = 0; kk < K; kk += 128) {
    // prefetch next B tile (gfx1250 global_prefetch_b8)
    if (kk + 128 < K) __builtin_prefetch(sBp + kk + 128, 0, 1);

    // ---- stage B tile (32 rows x 128 K) as bf16 in LDS
#pragma unroll
    for (int q = 0; q < 4; ++q) {
      const float4 f = *(const float4*)(sBp + kk + q * 4);
      ushort4 u;
      u.x = f2bf_rne(f.x); u.y = f2bf_rne(f.y);
      u.z = f2bf_rne(f.z); u.w = f2bf_rne(f.w);
      *(ushort4*)&ldsB[sn * 128 + sk0 + q * 4] = u;
    }
    __syncthreads();

#pragma unroll
    for (int j = 0; j < 4; ++j) {
      const int kb = kk + j * 32;
      // ---- A fragments (ISA 16-bit A layout), direct bf16 loads:
      // lanes 0-15: K = kb+0..7, kb+16..23 ; lanes 16-31: K = kb+8..15, kb+24..31
      union { uint4 q[2]; v16us v; } a0u, a1u, b0u, b1u;
      {
        const unsigned short* ap0 = A + (long)(mBase + l16) * lda + kb + half * 8;
        const unsigned short* ap1 = ap0 + (long)16 * lda;
        a0u.q[0] = *(const uint4*)(ap0);
        a0u.q[1] = *(const uint4*)(ap0 + 16);
        a1u.q[0] = *(const uint4*)(ap1);
        a1u.q[1] = *(const uint4*)(ap1 + 16);
      }
      // ---- B fragments: lane n=l16; lanes 0-15 K kb+0..7, kb+16..23 ; lanes 16-31 kb+8..15, kb+24..31
      {
        const uint4* lp0 = (const uint4*)&ldsB[l16 * 128 + j * 32 + half * 8];
        const uint4* lp1 = (const uint4*)&ldsB[(16 + l16) * 128 + j * 32 + half * 8];
        b0u.q[0] = lp0[0]; b0u.q[1] = lp0[2];
        b1u.q[0] = lp1[0]; b1u.q[1] = lp1[2];
      }
      const v16bf af0 = __builtin_bit_cast(v16bf, a0u.v);
      const v16bf af1 = __builtin_bit_cast(v16bf, a1u.v);
      const v16bf bf0 = __builtin_bit_cast(v16bf, b0u.v);
      const v16bf bf1 = __builtin_bit_cast(v16bf, b1u.v);
      acc00 = __builtin_amdgcn_wmma_f32_16x16x32_bf16(false, af0, false, bf0,
                                                      (short)0, acc00, false, false);
      acc01 = __builtin_amdgcn_wmma_f32_16x16x32_bf16(false, af0, false, bf1,
                                                      (short)0, acc01, false, false);
      acc10 = __builtin_amdgcn_wmma_f32_16x16x32_bf16(false, af1, false, bf0,
                                                      (short)0, acc10, false, false);
      acc11 = __builtin_amdgcn_wmma_f32_16x16x32_bf16(false, af1, false, bf1,
                                                      (short)0, acc11, false, false);
    }
    __syncthreads();
  }

  // ---- store C (ISA C/D layout: VGPR r -> M = r + 8*half, N = l16)
#pragma unroll
  for (int r = 0; r < 8; ++r) {
    const int row0 = mBase + half * 8 + r;
    const int row1 = row0 + 16;
    const int col0 = nTile + l16;
    const int col1 = col0 + 16;
    if (col0 < N) {
      C[(long)row0 * ldc + col0] = acc00[r];
      C[(long)row1 * ldc + col0] = acc10[r];
      if (WB16) {
        Cb[(long)row0 * ldc + col0] = f2bf_rne(acc00[r]);
        Cb[(long)row1 * ldc + col0] = f2bf_rne(acc10[r]);
      }
    }
    if (col1 < N) {
      C[(long)row0 * ldc + col1] = acc01[r];
      C[(long)row1 * ldc + col1] = acc11[r];
      if (WB16) {
        Cb[(long)row0 * ldc + col1] = f2bf_rne(acc01[r]);
        Cb[(long)row1 * ldc + col1] = f2bf_rne(acc11[r]);
      }
    }
  }
}

// ---------------------------------------------------------------------------
// x = emb[input_id]  fp32 -> d_out, bf16 shadow -> ws   (4 floats/thread)
// ---------------------------------------------------------------------------
__global__ __launch_bounds__(256) void gather_rows(
    const int* __restrict__ ids, const float* __restrict__ emb,
    float* __restrict__ x, unsigned short* __restrict__ xb) {
  const int s = blockIdx.x;
  const float4 v = ((const float4*)(emb + (long)ids[s] * DMODEL))[threadIdx.x];
  ((float4*)(x + (long)s * DMODEL))[threadIdx.x] = v;
  ushort4 u;
  u.x = f2bf_rne(v.x); u.y = f2bf_rne(v.y);
  u.z = f2bf_rne(v.z); u.w = f2bf_rne(v.w);
  ((ushort4*)(xb + (long)s * DMODEL))[threadIdx.x] = u;
}

// fp32 -> bf16 convert (h_state)
__global__ __launch_bounds__(256) void conv_bf16(
    const float* __restrict__ src, unsigned short* __restrict__ dst) {
  const int i = blockIdx.x * 256 + threadIdx.x;
  dst[i] = f2bf_rne(src[i]);
}

// wm_new = carry * sigmoid(P1) * wm_state + x   (bf16 only; used only as A)
__global__ __launch_bounds__(256) void ew_wm(
    const float* __restrict__ P1, const float* __restrict__ wm_state,
    const float* __restrict__ x, const unsigned char* __restrict__ reset,
    unsigned short* __restrict__ wm_new_b) {
  const int i = blockIdx.x * 256 + threadIdx.x;
  const int s = i >> 10;
  const float carry = reset[s] ? 0.0f : 1.0f;
  wm_new_b[i] = f2bf_rne(carry * sigmoidf_(P1[i]) * wm_state[i] + x[i]);
}

// y = tanh(p)  fp32 -> d_out, bf16 shadow -> ws
__global__ __launch_bounds__(256) void ew_tanh(
    const float* __restrict__ p, float* __restrict__ y,
    unsigned short* __restrict__ yb) {
  const int i = blockIdx.x * 256 + threadIdx.x;
  const float t = tanhf(p[i]);
  y[i] = t;
  yb[i] = f2bf_rne(t);
}

// g = sigmoid(gpre + surprise*w_s + b_g); cur = carry*((1-g)*h_old + g*tanh(a))
// cur consumed only as a GEMM A operand -> bf16 only.
__global__ __launch_bounds__(256) void ew_gate(
    const float* __restrict__ gpre, const float* __restrict__ a,
    const float* __restrict__ h_old, const float* __restrict__ surprise,
    const float* __restrict__ w_s, const float* __restrict__ b_g,
    const unsigned char* __restrict__ reset,
    unsigned short* __restrict__ cur_out_b) {
  const int i  = blockIdx.x * 256 + threadIdx.x;  // over 256*1024
  const int s  = i >> 10;
  const int bd = i & 1023;  // b*128 + dh
  const float g = sigmoidf_(gpre[i] + surprise[s] * w_s[bd] + b_g[bd]);
  const float carry = reset[s] ? 0.0f : 1.0f;
  cur_out_b[i] = f2bf_rne(carry * ((1.0f - g) * h_old[i] + g * tanhf(a[i])));
}

// ---------------------------------------------------------------------------
extern "C" void kernel_launch(void* const* d_in, const int* in_sizes, int n_in,
                              void* d_out, int out_size, void* d_ws,
                              size_t ws_size, hipStream_t stream) {
  (void)in_sizes; (void)n_in; (void)out_size; (void)ws_size;

  const int*   input_id = (const int*)d_in[0];
  const unsigned char* reset = (const unsigned char*)d_in[1];  // jax bool (1B)
  const float* surprise = (const float*)d_in[2];
  const float* wm_state = (const float*)d_in[3];
  const float* h_state  = (const float*)d_in[4];
  const float* emb      = (const float*)d_in[5];
  const float* W_in     = (const float*)d_in[6];
  const float* W_alpha  = (const float*)d_in[7];
  const float* W_wm_out = (const float*)d_in[8];
  const float* W_x      = (const float*)d_in[9];
  const float* W_h      = (const float*)d_in[10];
  const float* W_wml    = (const float*)d_in[11];
  const float* W_g      = (const float*)d_in[12];
  const float* w_s      = (const float*)d_in[13];
  const float* b_g      = (const float*)d_in[14];

  float* out    = (float*)d_out;
  float* logits = out;                              // 256*50257
  float* x_out  = out + (size_t)BS * VOCAB;         // 256*1024
  float* y_out  = x_out + (size_t)BS * DMODEL;      // 256*1024

  char* ws = (char*)d_ws;
  const size_t MB = 1u << 20;
  float*          P      = (float*)(ws + 0 * MB);            // fp32 256x1024
  unsigned short* wm_b   = (unsigned short*)(ws + 1 * MB);   // bf16 256x1024
  float*          a_buf  = (float*)(ws + 2 * MB);            // fp32 (256,8,128)
  float*          gpre   = (float*)(ws + 3 * MB);            // fp32 256x1024
  unsigned short* xb     = (unsigned short*)(ws + 4 * MB);   // bf16 x
  unsigned short* yb     = (unsigned short*)(ws + 4 * MB + 512 * 1024);
  unsigned short* cur0b  = (unsigned short*)(ws + 5 * MB);
  unsigned short* cur1b  = (unsigned short*)(ws + 5 * MB + 512 * 1024);
  unsigned short* a_bf   = (unsigned short*)(ws + 6 * MB);
  unsigned short* hsb    = (unsigned short*)(ws + 6 * MB + 512 * 1024);  // 1MB
  float*          cdump  = (float*)(ws + 8 * MB);            // fp32 dump

  const dim3 blk(256);
  const dim3 gD(DMODEL / 32, 1);          // dense 256x1024 GEMMs
  const dim3 gL(DH / 32, NBLK);           // batched per-block einsums
  const dim3 gV((VOCAB + 31) / 32, 1);    // vocab GEMM (tail guarded)

  // x = emb[input_id] (+ bf16 shadow); h_state bf16 shadow
  gather_rows<<<BS, 256, 0, stream>>>(input_id, emb, x_out, xb);
  conv_bf16<<<(NLAYER * BS * NBLK * DH) / 256, 256, 0, stream>>>(h_state, hsb);

  // P1 = x @ W_alpha^T ; x_proj = x @ W_in^T (bf16 shadow -> cur0b)
  wmma_gemm_nt<false, false><<<gD, blk, 0, stream>>>(xb, W_alpha, P, nullptr,
      DMODEL, DMODEL, DMODEL, DMODEL, DMODEL, 0, 0, 0);
  wmma_gemm_nt<false, true><<<gD, blk, 0, stream>>>(xb, W_in, cdump, cur0b,
      DMODEL, DMODEL, DMODEL, DMODEL, DMODEL, 0, 0, 0);
  // wm_new (bf16) = carry * sigmoid(P1) * wm_state + x
  ew_wm<<<(BS * DMODEL) / 256, 256, 0, stream>>>(P, wm_state, x_out, reset, wm_b);
  // y_wm = tanh(wm_new @ W_wm_out^T)
  wmma_gemm_nt<false, false><<<gD, blk, 0, stream>>>(wm_b, W_wm_out, P, nullptr,
      DMODEL, DMODEL, DMODEL, DMODEL, DMODEL, 0, 0, 0);
  ew_tanh<<<(BS * DMODEL) / 256, 256, 0, stream>>>(P, y_out, yb);

  unsigned short* curInB = cur0b;
  unsigned short* curOutB = cur1b;
  for (int l = 0; l < NLAYER; ++l) {
    const float* Wx  = W_x  + (size_t)l * NBLK * DH * DH;
    const float* Wh  = W_h  + (size_t)l * NBLK * DH * DH;
    const float* Wml = W_wml + (size_t)l * NBLK * DH * DMODEL;
    const float* Wg  = W_g  + (size_t)l * NBLK * DH * DH;
    const float*          hs   = h_state + (size_t)l * BS * NBLK * DH;
    const unsigned short* hsbl = hsb + (size_t)l * BS * NBLK * DH;

    // a = cur@Wx^T + h@Wh^T + y_wm@Wml^T (batched over b; last writes bf16 a)
    wmma_gemm_nt<false, false><<<gL, blk, 0, stream>>>(curInB, Wx, a_buf,
        nullptr, DH, DH, DMODEL, DH, DMODEL, DH, (long)DH * DH, DH);
    wmma_gemm_nt<true, false><<<gL, blk, 0, stream>>>(hsbl, Wh, a_buf,
        nullptr, DH, DH, DMODEL, DH, DMODEL, DH, (long)DH * DH, DH);
    wmma_gemm_nt<true, true><<<gL, blk, 0, stream>>>(yb, Wml, a_buf,
        a_bf, DH, DMODEL, DMODEL, DMODEL, DMODEL, 0, (long)DH * DMODEL, DH);
    // gpre = a @ Wg^T
    wmma_gemm_nt<false, false><<<gL, blk, 0, stream>>>(a_bf, Wg, gpre,
        nullptr, DH, DH, DMODEL, DH, DMODEL, DH, (long)DH * DH, DH);
    // gate + state update -> bf16 cur
    ew_gate<<<(BS * DMODEL) / 256, 256, 0, stream>>>(
        gpre, a_buf, hs, surprise, w_s + l * NBLK * DH, b_g + l * NBLK * DH,
        reset, curOutB);
    unsigned short* t = curInB; curInB = curOutB; curOutB = t;
  }

  // logits = h_final @ emb^T  (dominant kernel: single-pass 206 MB emb stream)
  wmma_gemm_nt<false, false><<<gV, blk, 0, stream>>>(curInB, emb, logits,
      nullptr, VOCAB, DMODEL, DMODEL, DMODEL, VOCAB, 0, 0, 0);
}
